// GraphAttentionLayer_18915035971929
// MI455X (gfx1250) — matrix operand order repaired and checked by reading, not compiled, not run on hardware
//
#include <hip/hip_runtime.h>
#include <hip/hip_bf16.h>
#include <stdint.h>

// ---------------------------------------------------------------------------
// GraphAttentionLayer for MI455X (gfx1250, wave32, WMMA)
// B=8, U=1024, N=2048, Din=128, H=4, HD=32
// ---------------------------------------------------------------------------

typedef __attribute__((ext_vector_type(2)))  float    v2f;
typedef __attribute__((ext_vector_type(8)))  float    v8f;
typedef __attribute__((ext_vector_type(16))) __bf16   v16bf;
typedef __attribute__((ext_vector_type(4)))  unsigned v4u;

#define ALPHA  0.2f
#define BATCH  8
#define NUSER  1024
#define NNEWS  2048
#define DIN    128
#define NH     4
#define HD     32
#define ADJDIM 3072

#if defined(__has_builtin)
#  if __has_builtin(__builtin_amdgcn_wmma_f32_16x16x4_f32)
#    define HAVE_WMMA_F32 1
#  else
#    define HAVE_WMMA_F32 0
#  endif
#else
#  define HAVE_WMMA_F32 0
#endif

__device__ __forceinline__ unsigned short f2bf_bits(float x) {
  unsigned u = __builtin_bit_cast(unsigned, x);
  unsigned r = u + 0x7FFFu + ((u >> 16) & 1u);   // round-to-nearest-even
  return (unsigned short)(r >> 16);
}

union AOp { unsigned short s[16]; v16bf v; };
union BOp { v4u q[2];             v16bf v; };

// ---------------------------------------------------------------------------
// Kernel 1: projections q_user / k_news via f32 WMMA, e-vectors, and
// bf16 V-matrices stored pre-swizzled in WMMA B-operand lane layout.
// Block = 128 threads (4 waves = 4 heads). Grid = B * ((U+N)/16).
// ---------------------------------------------------------------------------
__global__ __launch_bounds__(128) void proj_kernel(
    const float* __restrict__ Xu, const float* __restrict__ Xn,
    const float* __restrict__ Wu, const float* __restrict__ Wn,
    const float* __restrict__ Aun, const float* __restrict__ Anu,
    float* __restrict__ eq_un, float* __restrict__ ek_nu,
    float* __restrict__ eq_nu, float* __restrict__ ek_un,
    char* __restrict__ bswz_u, char* __restrict__ bswz_n)
{
  __shared__ float Xs[16 * 130];        // 16 rows x Din, padded stride
  __shared__ float Cs[NH][16 * 33];     // per-wave 16x32 result, padded

  const int T  = threadIdx.x;
  const int L  = T & 31;                // lane
  const int h  = T >> 5;                // wave id == head
  const int b  = blockIdx.x / 192;
  const int tr = blockIdx.x % 192;
  const int side = (tr < 64) ? 0 : 1;   // 0 = user rows, 1 = news rows
  const int r0   = side ? (tr - 64) * 16 : tr * 16;
  const int Rows = side ? NNEWS : NUSER;
  const float* X = side ? (Xn + ((size_t)b * NNEWS + r0) * DIN)
                        : (Xu + ((size_t)b * NUSER + r0) * DIN);

  // stage X tile (16 x 128), coalesced; features are read exactly once
  for (int j = 0; j < 16; ++j)
    Xs[j * 130 + T] = __builtin_nontemporal_load(&X[(size_t)j * DIN + T]);
  __syncthreads();

  const float* Wg = (side ? Wn : Wu) + h * DIN * HD;   // [Din][HD]
  const int g  = L >> 4;                // lane half
  const int lN = L & 15;

  v8f c0 = {}; v8f c1 = {};
#if HAVE_WMMA_F32
  // A layout (f32 16x4): lane M = L%16, element v -> K = 2g+v
  // B layout (f32 4x16): lane N = L%16, element v -> K = 2g+v
  const float* xrow = &Xs[(L & 15) * 130];
  for (int d0 = 0; d0 < DIN; d0 += 4) {
    v2f a;  a.x  = xrow[d0 + 2 * g];          a.y  = xrow[d0 + 2 * g + 1];
    v2f b0; b0.x = Wg[(d0 + 2 * g) * HD + lN];
            b0.y = Wg[(d0 + 2 * g + 1) * HD + lN];
    v2f b1; b1.x = Wg[(d0 + 2 * g) * HD + 16 + lN];
            b1.y = Wg[(d0 + 2 * g + 1) * HD + 16 + lN];
    c0 = __builtin_amdgcn_wmma_f32_16x16x4_f32(false, a, false, b0, (short)0, c0, false, false);
    c1 = __builtin_amdgcn_wmma_f32_16x16x4_f32(false, a, false, b1, (short)0, c1, false, false);
  }
#else
  // scalar fallback, same C layout (M = j + 8g, N = lN)
  for (int j = 0; j < 8; ++j) {
    const int M = j + 8 * g;
    float s0 = 0.f, s1 = 0.f;
    for (int d = 0; d < DIN; ++d) {
      const float xv = Xs[M * 130 + d];
      s0 += xv * Wg[d * HD + lN];
      s1 += xv * Wg[d * HD + 16 + lN];
    }
    c0[j] = s0; c1[j] = s1;
  }
#endif

  // spill C tile (16 rows x 32 cols) to LDS: C element (M=j+8g, N=lane%16)
  float* Cw = Cs[h];
#pragma unroll
  for (int j = 0; j < 8; ++j) {
    Cw[(j + 8 * g) * 33 + lN]      = c0[j];
    Cw[(j + 8 * g) * 33 + 16 + lN] = c1[j];
  }
  __syncthreads();

  // e-scalars: 16 rows x 2 dot products; lane L -> row L%16, selector L/16
  {
    const int r = L & 15;
    const float* av;
    float* dst;
    if (side == 0) { av = g ? (Anu + h * 64 + 32) : (Aun + h * 64);
                     dst = g ? ek_nu : eq_un; }
    else           { av = g ? (Aun + h * 64 + 32) : (Anu + h * 64);
                     dst = g ? ek_un : eq_nu; }
    float e = 0.f;
#pragma unroll
    for (int k = 0; k < HD; ++k) e += Cw[r * 33 + k] * av[k];
    dst[(size_t)(b * NH + h) * Rows + r0 + r] = e;
  }

  // bf16 V store in WMMA B-operand swizzled layout:
  // reader lane Lr (group gr=Lr/16, col N=Lr%16) element e -> K = gr*16 + e.
  // This 16-row tile covers K-group gr = halfk of chunk nc = r0/32.
  {
    char* bswz = side ? bswz_n : bswz_u;
    const int NC    = side ? (NNEWS / 32) : (NUSER / 32);
    const int nc    = r0 >> 5;
    const int halfk = (r0 >> 4) & 1;
    const int t     = g;               // which HD tile (cols 0-15 / 16-31)
    unsigned pk[8];
#pragma unroll
    for (int j = 0; j < 8; ++j) {
      const unsigned lo = f2bf_bits(Cw[(2 * j) * 33 + t * 16 + lN]);
      const unsigned hi = f2bf_bits(Cw[(2 * j + 1) * 33 + t * 16 + lN]);
      pk[j] = lo | (hi << 16);
    }
    const int lanep = halfk * 16 + lN;
    const size_t off =
        ((((size_t)(b * NH + h) * NC + nc) * 2 + t) * 32 + lanep) * 32;  // bytes
    v4u* dp = (v4u*)(bswz + off);
    v4u lo4; lo4.x = pk[0]; lo4.y = pk[1]; lo4.z = pk[2]; lo4.w = pk[3];
    v4u hi4; hi4.x = pk[4]; hi4.y = pk[5]; hi4.z = pk[6]; hi4.w = pk[7];
    dp[0] = lo4; dp[1] = hi4;   // re-read by kernel 2 -> keep regular (L2)
  }
}

// ---------------------------------------------------------------------------
// Kernel 2: fused masked-softmax-attention x V, single pass (no max-sub:
// softmax is shift invariant; scores are bounded so f32 exp is safe).
// Block = 128 (4 waves = 4 heads sharing the adjacency tile in LDS).
// Wave: 16 rows x HD=32 output. Mask staged 16x128 per barrier (b128 loads,
// non-temporal: adjacency is read-once and must not evict L2-resident V/e).
// ---------------------------------------------------------------------------
__global__ __launch_bounds__(128) void attn_kernel(
    const int* __restrict__ adj,
    const float* __restrict__ eqv, const float* __restrict__ ekv,
    const char* __restrict__ bswz, const float* __restrict__ bias,
    float* __restrict__ outp, int side)
{
  __shared__ v4u maskQ[16 * 33];                 // 16 rows x 128 ints, padded
  int* maskI = (int*)maskQ;                      // row stride = 132 ints

  const int T  = threadIdx.x;
  const int L  = T & 31;
  const int h  = T >> 5;
  const int RT = side ? (NNEWS / 16) : (NUSER / 16);
  const int b  = blockIdx.x / RT;
  const int rt = blockIdx.x % RT;
  const int r0 = rt * 16;
  const int Rows  = side ? NNEWS : NUSER;
  const int Ncols = side ? NUSER : NNEWS;
  const int NC32  = Ncols / 32;                  // WMMA K-chunks
  const int NC128 = Ncols / 128;                 // staging chunks
  const int g  = L >> 4;
  const int r  = L & 15;
  const int lN = L & 15;
  const int bh = b * NH + h;

  const int maskRow0 = side ? (NUSER + r0) : r0;
  const int colOff   = side ? 0 : NUSER;
  const int* mbase = adj + (size_t)b * ADJDIM * ADJDIM
                         + (size_t)maskRow0 * ADJDIM + colOff;

  const float  eqr = eqv[(size_t)bh * Rows + r0 + r];
  const float* ekb = ekv + (size_t)bh * Ncols;
  const char*  bbase = bswz + (((size_t)bh * NC32) * 2 * 32 + L) * 32;

  v8f c0 = {}; v8f c1 = {};
  float lsum = 0.f;

  for (int nc = 0; nc < NC128; ++nc) {
    // stage mask 16x128 ints as 512 b128 loads by 128 threads (coalesced, NT)
#pragma unroll
    for (int j = 0; j < 4; ++j) {
      const int flatq = j * 128 + T;
      const int row = flatq >> 5, qc = flatq & 31;
      const v4u* src =
          (const v4u*)(mbase + (size_t)row * ADJDIM + nc * 128 + qc * 4);
      maskQ[row * 33 + qc] = __builtin_nontemporal_load(src);
    }
    __syncthreads();

#pragma unroll
    for (int sc = 0; sc < 4; ++sc) {
      const int nb = sc * 32;                    // sub-chunk column base
      // e_k for this lane's 16 A-elements: n_loc in {16g..16g+15}
      const float4* p0 = (const float4*)(ekb + nc * 128 + nb + 16 * g);
      const float4* p1 = (const float4*)(ekb + nc * 128 + nb + 16 * g + 8);
      const float4 q0 = p0[0], q1 = p0[1], q2 = p1[0], q3 = p1[1];
      const float ekr[16] = {q0.x, q0.y, q0.z, q0.w, q1.x, q1.y, q1.z, q1.w,
                             q2.x, q2.y, q2.z, q2.w, q3.x, q3.y, q3.z, q3.w};

      // build unnormalized-prob A tile (16x32 bf16; A layout: row M = L%16,
      // element e -> K = 16g + e)
      AOp A;
#pragma unroll
      for (int e = 0; e < 16; ++e) {
        const int n_loc = 16 * g + e;
        const int m = maskI[r * 132 + nb + n_loc];
        float s = eqr + ekr[e];
        s = s > 0.f ? s : ALPHA * s;                 // leaky relu
        const float ex = (m > 0) ? __expf(s) : 0.f;  // masked exp
        lsum += ex;
        A.s[e] = f2bf_bits(ex);
      }

      // B tiles: two 128-bit loads each from pre-swizzled layout (L2-hot)
      const int c32 = nc * 4 + sc;
      const char* bp = bbase + (size_t)c32 * (2 * 32 * 32);
      BOp B0, B1;
      B0.q[0] = ((const v4u*)bp)[0];
      B0.q[1] = ((const v4u*)bp)[1];
      B1.q[0] = ((const v4u*)(bp + 32 * 32))[0];
      B1.q[1] = ((const v4u*)(bp + 32 * 32))[1];

      c0 = __builtin_amdgcn_wmma_f32_16x16x32_bf16(false, A.v, false, B0.v, (short)0, c0, false, false);
      c1 = __builtin_amdgcn_wmma_f32_16x16x32_bf16(false, A.v, false, B1.v, (short)0, c1, false, false);
    }
    __syncthreads();   // before next staging overwrites the mask tile
  }

  // per-row softmax denominator: lanes L and L^16 hold the two halves
  const float lt = lsum + __shfl_xor(lsum, 16);

  float* outT = outp + (side ? (size_t)BATCH * NUSER * NH * HD +
                               ((size_t)bh * NNEWS + r0) * HD
                             : ((size_t)bh * NUSER + r0) * HD);
#pragma unroll
  for (int j = 0; j < 8; ++j) {
    const int M = j + 8 * g;                  // C row for this VGPR
    const float lj  = __shfl(lt, j + 8 * g);  // lane M holds row M's total
    const float inv = lj > 0.f ? 1.f / lj : 0.f;
    const int brow = ((r0 + M) & 3) * 32;     // bias idx = (row%4)*32 + hd
    const float o0 = c0[j] * inv + bias[brow + lN];
    const float o1 = c1[j] * inv + bias[brow + 16 + lN];
    __builtin_nontemporal_store(o0 > 0.f ? o0 : 0.f, &outT[(size_t)M * HD + lN]);
    __builtin_nontemporal_store(o1 > 0.f ? o1 : 0.f, &outT[(size_t)M * HD + 16 + lN]);
  }
}

// ---------------------------------------------------------------------------
extern "C" void kernel_launch(void* const* d_in, const int* in_sizes, int n_in,
                              void* d_out, int out_size, void* d_ws, size_t ws_size,
                              hipStream_t stream) {
  const float* Xu  = (const float*)d_in[0];
  const float* Xn  = (const float*)d_in[1];
  const int*   adj = (const int*)  d_in[2];
  const float* Wu  = (const float*)d_in[3];
  const float* Wn  = (const float*)d_in[4];
  const float* Aun = (const float*)d_in[5];
  const float* Anu = (const float*)d_in[6];
  const float* bu  = (const float*)d_in[7];
  const float* bn  = (const float*)d_in[8];

  char* ws = (char*)d_ws;
  float* eq_un = (float*)(ws + 0);        // B*H*U f32
  float* ek_nu = (float*)(ws + 131072);   // B*H*U f32
  float* ek_un = (float*)(ws + 262144);   // B*H*N f32
  float* eq_nu = (float*)(ws + 524288);   // B*H*N f32
  char*  bswz_u = ws + 786432;            // B*H*U*HD bf16  (2 MB)
  char*  bswz_n = ws + 786432 + 2097152;  // B*H*N*HD bf16  (4 MB)

  proj_kernel<<<BATCH * 192, 128, 0, stream>>>(
      Xu, Xn, Wu, Wn, Aun, Anu, eq_un, ek_nu, eq_nu, ek_un, bswz_u, bswz_n);

  // user rows attend over news (V = k_news)
  attn_kernel<<<BATCH * (NUSER / 16), 128, 0, stream>>>(
      adj, eq_un, ek_un, bswz_n, bu, (float*)d_out, 0);
  // news rows attend over users (V = q_user)
  attn_kernel<<<BATCH * (NNEWS / 16), 128, 0, stream>>>(
      adj, eq_nu, ek_nu, bswz_u, bn, (float*)d_out, 1);
}
